// GIN_81896436400700
// MI455X (gfx1250) — matrix operand order repaired and checked by reading, not compiled, not run on hardware
//
#include <hip/hip_runtime.h>
#include <hip/hip_bf16.h>

typedef __attribute__((ext_vector_type(16))) _Float16     v16h;
typedef __attribute__((ext_vector_type(8)))  float        v8f;
typedef __attribute__((ext_vector_type(4)))  unsigned int u32x4;
typedef __attribute__((ext_vector_type(4)))  int          i32x4;
typedef __attribute__((ext_vector_type(4)))  float        f32x4;

union Frag { v16h h; u32x4 q[2]; };

#define AM_F32    0
#define AM_F16    1
#define AM_GATHER 2
#define EP_BN            0
#define EP_BIAS_RELU_F16 1
#define EP_BIAS_F32      2

// ---- gfx1250 async Global->LDS path (ASYNCcnt), guarded for compile safety ----
#if defined(__gfx1250__) && __has_builtin(__builtin_amdgcn_global_load_async_to_lds_b128)
#define USE_ASYNC_LDS 1
#else
#define USE_ASYNC_LDS 0
#endif

__device__ __forceinline__ void lds_async_b128(const void* g, void* l) {
#if USE_ASYNC_LDS
  __builtin_amdgcn_global_load_async_to_lds_b128(
      (__attribute__((address_space(1))) i32x4*)(void*)g,
      (__attribute__((address_space(3))) i32x4*)(void*)l, 0, 0);
#else
  (void)g; (void)l;
#endif
}

__device__ __forceinline__ void lds_wait_async() {
#if USE_ASYNC_LDS
#if __has_builtin(__builtin_amdgcn_s_wait_asynccnt)
  __builtin_amdgcn_s_wait_asynccnt(0);
#else
  asm volatile("s_wait_asynccnt 0x0" ::: "memory");
#endif
#endif
}

// ---------------------------------------------------------------------------
// Tiled WMMA GEMM: C[M,NC] = A[M,K] @ W[K,NC], W given transposed f16 Wt[NC][K].
// Block = 256 threads (8 waves). Block tile = 128 rows x 128 cols.
// Double-buffered LDS: stage chunk kc+1 (B via async Global->LDS when
// available) while computing chunk kc; one barrier per K-chunk.
// B fragments double-buffered in registers so ds_load of frag j+1 overlaps
// the WMMA consuming frag j.
// ---------------------------------------------------------------------------
template<int AMODE, int EPI>
__global__ __launch_bounds__(256) void gemm_wmma(
    const void* __restrict__ Aptr,
    const int*  __restrict__ uIdx, const int* __restrict__ vIdx,
    const _Float16* __restrict__ Wt,     // [NC_total][K]
    const float* __restrict__ bias,      // [NC_total] or null
    void* __restrict__ Cptr,
    float* __restrict__ colsum, float* __restrict__ colsumsq,
    int M, int K, int NC)
{
  __shared__ __attribute__((aligned(16))) _Float16 As[2][128][72];
  __shared__ __attribute__((aligned(16))) _Float16 Bs[2][128][72];

  const int tid    = threadIdx.x;
  const int wave   = tid >> 5;
  const int lane   = tid & 31;
  const int halfId = lane >> 4;
  const int m16    = lane & 15;
  const int row0   = blockIdx.x * 128;
  const int n0     = blockIdx.y * 128;

  v8f acc[8] = {};
  const int nk = K >> 6;

  auto stage = [&](int buf, int k0) {
    // ---- stage B: 128 cols x 64 K-halves (pure f16 copy -> async-capable) ----
    #pragma unroll
    for (int t = tid; t < 1024; t += 256) {
      int n  = t >> 3;
      int c8 = (t & 7) << 3;
      const _Float16* g = &Wt[(size_t)(n0 + n) * K + k0 + c8];
#if USE_ASYNC_LDS
      lds_async_b128(g, &Bs[buf][n][c8]);
#else
      *(u32x4*)&Bs[buf][n][c8] = *(const u32x4*)g;
#endif
    }
    // ---- stage A: 128 rows x 64 K-halves (convert / zero-fill guard rows) ----
    if (AMODE == AM_F16) {
      const _Float16* A = (const _Float16*)Aptr;
      #pragma unroll
      for (int t = tid; t < 1024; t += 256) {
        int r = t >> 3;
        int c8 = (t & 7) << 3;
        int row = row0 + r;
        u32x4 vv = {0u, 0u, 0u, 0u};
        if (row < M) vv = *(const u32x4*)&A[(size_t)row * K + k0 + c8];
        *(u32x4*)&As[buf][r][c8] = vv;
      }
    } else {
      #pragma unroll
      for (int t = tid; t < 2048; t += 256) {
        int r  = t >> 4;
        int c4 = (t & 15) << 2;
        int row = row0 + r;
        f32x4 vv = {0.f, 0.f, 0.f, 0.f};
        if (row < M) {
          if (AMODE == AM_F32) {
            const float* A = (const float*)Aptr;
            vv = *(const f32x4*)&A[(size_t)row * K + k0 + c4];
            if (k0 + 64 < K)
              __builtin_prefetch(&A[(size_t)row * K + k0 + 64 + c4], 0, 1);
          } else { // AM_GATHER: row e = [hsum[u[e]] | hsum[v[e]]], stride 512
            const float* Hs = (const float*)Aptr;
            int node = (k0 < 512) ? uIdx[row] : vIdx[row];
            int kk   = (k0 < 512) ? (k0 + c4) : (k0 + c4 - 512);
            vv = *(const f32x4*)&Hs[(size_t)node * 512 + kk];
          }
        }
        As[buf][r][c4 + 0] = (_Float16)vv[0];
        As[buf][r][c4 + 1] = (_Float16)vv[1];
        As[buf][r][c4 + 2] = (_Float16)vv[2];
        As[buf][r][c4 + 3] = (_Float16)vv[3];
      }
    }
  };

  stage(0, 0);
  lds_wait_async();
  __syncthreads();

  int buf = 0;
  for (int kc = 0; kc < nk; ++kc) {
    if (kc + 1 < nk) stage(buf ^ 1, (kc + 1) << 6);

    // ---- 2 K-steps of 32, 8 N-tiles each: 16 WMMAs per chunk ----
    #pragma unroll
    for (int kt = 0; kt < 2; ++kt) {
      Frag a;
      const int arow = (wave << 4) + m16;
      const int ak   = (kt << 5) + (halfId << 4);     // K 0-15 / 16-31 per lane-half
      a.q[0] = *(const u32x4*)&As[buf][arow][ak];     // + K 8-15 / 24-31 below
      a.q[1] = *(const u32x4*)&As[buf][arow][ak + 8];

      const int bk = (kt << 5) + (halfId << 4);       // K 0-15 / 16-31 per lane-half
      Frag b[2];                                      // register double-buffer
      b[0].q[0] = *(const u32x4*)&Bs[buf][m16][bk];
      b[0].q[1] = *(const u32x4*)&Bs[buf][m16][bk + 8];
      #pragma unroll
      for (int j = 0; j < 8; ++j) {
        if (j < 7) {
          const int brow = ((j + 1) << 4) + m16;
          b[(j + 1) & 1].q[0] = *(const u32x4*)&Bs[buf][brow][bk];
          b[(j + 1) & 1].q[1] = *(const u32x4*)&Bs[buf][brow][bk + 8];
        }
        acc[j] = __builtin_amdgcn_wmma_f32_16x16x32_f16(
            false, a.h, false, b[j & 1].h, (short)0, acc[j], false, false);
      }
    }

    lds_wait_async();   // next-chunk async B loads complete before barrier
    __syncthreads();
    buf ^= 1;
  }

  // ---- epilogue ----
  const int rbase = row0 + (wave << 4) + (halfId << 3);   // VGPR r -> M = rbase + r
  #pragma unroll
  for (int j = 0; j < 8; ++j) {
    const int col = n0 + (j << 4) + m16;
    const float bval = (EPI != EP_BN) ? bias[col] : 0.0f;
    float s = 0.f, sq = 0.f;
    #pragma unroll
    for (int r = 0; r < 8; ++r) {
      const int row = rbase + r;
      float cv = acc[j][r];
      if (EPI == EP_BN) {
        s += cv; sq += cv * cv;
        if (row < M) ((float*)Cptr)[(size_t)row * NC + col] = cv;
      } else if (EPI == EP_BIAS_RELU_F16) {
        cv += bval; cv = cv > 0.f ? cv : 0.f;
        if (row < M) ((_Float16*)Cptr)[(size_t)row * NC + col] = (_Float16)cv;
      } else {
        cv += bval;
        if (row < M) ((float*)Cptr)[(size_t)row * NC + col] = cv;
      }
    }
    if (EPI == EP_BN) {          // zero-padded rows contribute 0 -> stats safe
      atomicAdd(&colsum[col], s);
      atomicAdd(&colsumsq[col], sq);
    }
  }
}

// ---------------------------------------------------------------------------
// Elementwise / scatter kernels
// ---------------------------------------------------------------------------
__global__ void k_transpose_cvt(const float* __restrict__ in, _Float16* __restrict__ out,
                                int K, int NC) {
  int t = blockIdx.x * 256 + threadIdx.x;
  if (t >= K * NC) return;
  int n = t / K, k = t % K;
  out[t] = (_Float16)in[(size_t)k * NC + n];
}

__global__ void k_gather_init(const int* __restrict__ ids, const float* __restrict__ emb,
                              float* __restrict__ H, float* __restrict__ HS, int total4) {
  int t = blockIdx.x * 256 + threadIdx.x;
  if (t >= total4) return;
  int n = t >> 7, c4 = (t & 127) << 2;
  f32x4 vv = *(const f32x4*)&emb[(size_t)ids[n] * 512 + c4];
  *(f32x4*)&H [(size_t)n * 512 + c4] = vv;
  *(f32x4*)&HS[(size_t)n * 512 + c4] = vv;
}

__global__ void k_scale2(const float* __restrict__ H, float* __restrict__ Z, int total4) {
  int t = blockIdx.x * 256 + threadIdx.x;
  if (t >= total4) return;
  f32x4 vv = *(const f32x4*)&H[(size_t)t * 4];
  *(f32x4*)&Z[(size_t)t * 4] = vv + vv;
}

__global__ void k_edge_agg(const int* __restrict__ u, const int* __restrict__ v,
                           const float* __restrict__ H, float* __restrict__ Z, int E) {
  int t = blockIdx.x * 256 + threadIdx.x;
  if (t >= E * 128) return;               // 2 directions x 64 groups of 8 floats
  int g = t >> 6, off = (t & 63) << 3;
  int e = g >> 1;
  int s, d;
  if (g & 1) { s = u[e]; d = v[e]; } else { s = v[e]; d = u[e]; }
  const float* hs = &H[(size_t)s * 512 + off];
  float*       zd = &Z[(size_t)d * 512 + off];
  #pragma unroll
  for (int i = 0; i < 8; ++i) atomicAdd(&zd[i], hs[i]);
}

__global__ void k_zero_stats(float* a, float* b) {
  int t = blockIdx.x * 256 + threadIdx.x;
  if (t < 512) { a[t] = 0.f; b[t] = 0.f; }
}

__global__ void k_bn_relu_f16(const float* __restrict__ T, const float* __restrict__ cs,
                              const float* __restrict__ csq, const float* __restrict__ g,
                              const float* __restrict__ bt, _Float16* __restrict__ O,
                              float invN, int total4) {
  int t = blockIdx.x * 256 + threadIdx.x;
  if (t >= total4) return;
  int row = t >> 7, c4 = (t & 127) << 2;
  const float* tp = &T[(size_t)row * 512 + c4];
  _Float16*    op = &O[(size_t)row * 512 + c4];
  #pragma unroll
  for (int i = 0; i < 4; ++i) {
    int col = c4 + i;
    float mean = cs[col] * invN;
    float var  = csq[col] * invN - mean * mean;
    float val  = (tp[i] - mean) * rsqrtf(var + 1e-5f) * g[col] + bt[col];
    op[i] = (_Float16)(val > 0.f ? val : 0.f);
  }
}

__global__ void k_bn_relu_update(const float* __restrict__ T, const float* __restrict__ cs,
                                 const float* __restrict__ csq, const float* __restrict__ g,
                                 const float* __restrict__ bt, float* __restrict__ H,
                                 float* __restrict__ HS, float invN, int total4) {
  int t = blockIdx.x * 256 + threadIdx.x;
  if (t >= total4) return;
  int row = t >> 7, c4 = (t & 127) << 2;
  const float* tp = &T[(size_t)row * 512 + c4];
  #pragma unroll
  for (int i = 0; i < 4; ++i) {
    int col = c4 + i;
    size_t idx = (size_t)row * 512 + col;
    float mean = cs[col] * invN;
    float var  = csq[col] * invN - mean * mean;
    float val  = (tp[i] - mean) * rsqrtf(var + 1e-5f) * g[col] + bt[col];
    val = val > 0.f ? val : 0.f;
    H[idx]  = val;
    HS[idx] += val;
  }
}

// ---------------------------------------------------------------------------
extern "C" void kernel_launch(void* const* d_in, const int* in_sizes, int n_in,
                              void* d_out, int out_size, void* d_ws, size_t ws_size,
                              hipStream_t stream) {
  const int*   h_ids = (const int*)  d_in[0];
  const int*   u     = (const int*)  d_in[1];
  const int*   v     = (const int*)  d_in[2];
  const float* emb   = (const float*)d_in[3];
  const float* lin1  = (const float*)d_in[4];
  const float* lin2  = (const float*)d_in[5];
  const float* bn1_g = (const float*)d_in[6];
  const float* bn1_b = (const float*)d_in[7];
  const float* bn2_g = (const float*)d_in[8];
  const float* bn2_b = (const float*)d_in[9];
  const float* W1_w  = (const float*)d_in[10];
  const float* W1_b  = (const float*)d_in[11];
  const float* W2_w  = (const float*)d_in[12];
  const float* W2_b  = (const float*)d_in[13];

  const int N = in_sizes[0];
  const int E = in_sizes[1];
  const int D = 512, L = 4;
  const int DD = D * D;

  // ---- workspace carve-out ----
  char* ws = (char*)d_ws;
  size_t o = 0;
  auto carve = [&](size_t bytes) { size_t r = o; o += (bytes + 255) & ~(size_t)255; return r; };
  float*     H   = (float*)    (ws + carve((size_t)N * D * 4));
  float*     HS  = (float*)    (ws + carve((size_t)N * D * 4));
  float*     Z   = (float*)    (ws + carve((size_t)N * D * 4));   // also t2; T1E base
  _Float16*  A16 = (_Float16*) (ws + carve((size_t)N * D * 2));   // contiguous after Z
  float*     T   = (float*)    (ws + carve((size_t)N * D * 4));
  _Float16*  Wt1 = (_Float16*) (ws + carve((size_t)L * DD * 2));
  _Float16*  Wt2 = (_Float16*) (ws + carve((size_t)L * DD * 2));
  _Float16*  W1t = (_Float16*) (ws + carve((size_t)512 * 1024 * 2));
  _Float16*  W2t = (_Float16*) (ws + carve((size_t)128 * 512 * 2));
  float*     cs  = (float*)    (ws + carve(512 * 4));
  float*     csq = (float*)    (ws + carve(512 * 4));

  // ---- weight transpose + f16 convert ----
  for (int i = 0; i < L; ++i) {
    k_transpose_cvt<<<(DD + 255) / 256, 256, 0, stream>>>(lin1 + (size_t)i * DD, Wt1 + (size_t)i * DD, 512, 512);
    k_transpose_cvt<<<(DD + 255) / 256, 256, 0, stream>>>(lin2 + (size_t)i * DD, Wt2 + (size_t)i * DD, 512, 512);
  }
  k_transpose_cvt<<<(512 * 1024 + 255) / 256, 256, 0, stream>>>(W1_w, W1t, 1024, 512);
  k_transpose_cvt<<<(512 * 128  + 255) / 256, 256, 0, stream>>>(W2_w, W2t, 512, 128);

  const int tot4 = N * 128;                // N*D/4 float4 groups
  const float invN = 1.0f / (float)N;
  k_gather_init<<<(tot4 + 255) / 256, 256, 0, stream>>>(h_ids, emb, H, HS, tot4);

  dim3 gemmN((N + 127) / 128, 4);
  for (int i = 0; i < L; ++i) {
    k_scale2 <<<(tot4 + 255) / 256, 256, 0, stream>>>(H, Z, tot4);
    k_edge_agg<<<((size_t)E * 128 + 255) / 256, 256, 0, stream>>>(u, v, H, Z, E);

    k_zero_stats<<<2, 256, 0, stream>>>(cs, csq);
    gemm_wmma<AM_F32, EP_BN><<<gemmN, 256, 0, stream>>>(
        Z, nullptr, nullptr, Wt1 + (size_t)i * DD, nullptr, T, cs, csq, N, 512, 512);
    k_bn_relu_f16<<<(tot4 + 255) / 256, 256, 0, stream>>>(
        T, cs, csq, bn1_g + i * 512, bn1_b + i * 512, A16, invN, tot4);

    k_zero_stats<<<2, 256, 0, stream>>>(cs, csq);
    gemm_wmma<AM_F16, EP_BN><<<gemmN, 256, 0, stream>>>(
        A16, nullptr, nullptr, Wt2 + (size_t)i * DD, nullptr, Z, cs, csq, N, 512, 512);
    k_bn_relu_update<<<(tot4 + 255) / 256, 256, 0, stream>>>(
        Z, cs, csq, bn2_g + i * 512, bn2_b + i * 512, H, HS, invN, tot4);
  }

  // ---- decoder: relu(concat(h[u],h[v]) @ W1 + b1) @ W2 + b2 ----
  _Float16* T1E = (_Float16*)Z;            // overlays dead Z+A16 (E*512 f16)
  gemm_wmma<AM_GATHER, EP_BIAS_RELU_F16><<<dim3((E + 127) / 128, 4), 256, 0, stream>>>(
      HS, u, v, W1t, W1_b, T1E, nullptr, nullptr, E, 1024, 512);
  gemm_wmma<AM_F16, EP_BIAS_F32><<<dim3((E + 127) / 128, 1), 256, 0, stream>>>(
      T1E, nullptr, nullptr, W2t, W2_b, d_out, nullptr, nullptr, E, 512, 128);
}
